// FourierInterpolator_19052474925058
// MI455X (gfx1250) — matrix operand order repaired and checked by reading, not compiled, not run on hardware
//
#include <hip/hip_runtime.h>
#include <hip/hip_bf16.h>
#include <math.h>

// FourierInterpolator for MI455X (gfx1250, wave32, WMMA).
//
//  Kernel 1 (tiny): per-batch 64x64 DFT via LDS twiddle table, fold the 1/4096
//    scale in, store F^T (m-major, k-contiguous) as f16 planes: real, imag,
//    and real+imag (for the 3-multiply complex GEMM) in d_ws.
//  Kernel 2 (hot): per wave, 16 output points. E1 cos/sin A-fragments are
//    built from 10 sincos + complex rotations (frequency-shift identities
//    fk(e+8)=fk(e)+8, fk(ks=1)=fk(ks=0)-32); CS1 = C1+S1 via packed f16 adds.
//    F^T B-fragments come from LDS (staged via global_load_async_to_lds_b128).
//    Complex G = E1 @ F uses the Karatsuba 3-multiply form:
//      T1=C1*Fr, T2=S1*Fi, T3=(C1+S1)*(Fr+Fi);  Gr=T1-T2, Gi=T3-T1-T2
//    -> 24x v_wmma_f32_16x16x32_f16 per tile (vs 32 for the naive form).
//    The E2 row-dot reduction walks N-tiles in order {2,3,0,1} so the signed
//    frequency steps by +16 per tile: one start + one step sincos per row,
//    advanced by a 4-op complex multiply. Final 16-lane shfl_xor butterfly.
//
// Workspace: 3 * 32*64*64 f16 = 768 KB.

typedef __attribute__((ext_vector_type(16))) _Float16 v16h;
typedef __attribute__((ext_vector_type(8)))  float    v8f;

#define NB      32
#define S       64          // input side (both dims)
#define SS      (S * S)     // 4096
#define P       (128 * 128) // output points per batch
#define TWO_PI  6.28318530717958647692f

// ---------------------------------------------------------------------------
// Kernel 1: F = DFT2(y) / 4096, stored transposed as f16:
//           Frt[b][m][k], Fit[b][m][k], Fst = Frt + Fit.
// ---------------------------------------------------------------------------
__global__ __launch_bounds__(256) void FourierInterp_dft_kernel(
    const float* __restrict__ y, _Float16* __restrict__ Frt,
    _Float16* __restrict__ Fit, _Float16* __restrict__ Fst) {
  const int b = blockIdx.x;
  const int t = threadIdx.x;

  __shared__ float sy[SS];
  __shared__ float sTr[SS], sTi[SS];   // T[n1][m]
  __shared__ float twc[S], tws[S];     // exp(-2*pi*i*j/64)

  for (int i = t; i < SS; i += 256) sy[i] = y[b * SS + i];
  if (t < S) {
    float ang = -TWO_PI * (float)t / (float)S;
    twc[t] = __cosf(ang);
    tws[t] = __sinf(ang);
  }
  __syncthreads();

  // Stage 1: T[n1][m] = sum_n2 y[n1][n2] * W[(m*n2)%64]   (real input)
  for (int i = t; i < SS; i += 256) {
    const int n1 = i >> 6, m = i & 63;
    float ar = 0.f, ai = 0.f;
#pragma unroll 8
    for (int n2 = 0; n2 < S; ++n2) {
      const float v = sy[n1 * S + n2];
      const int idx = (m * n2) & 63;
      ar = fmaf(v, twc[idx], ar);
      ai = fmaf(v, tws[idx], ai);
    }
    sTr[i] = ar;
    sTi[i] = ai;
  }
  __syncthreads();

  // Stage 2: F[k][m] = sum_n1 W[(k*n1)%64] * T[n1][m]; write Ft[m][k]/4096
  for (int i = t; i < SS; i += 256) {
    const int k = i >> 6, m = i & 63;
    float ar = 0.f, ai = 0.f;
#pragma unroll 8
    for (int n1 = 0; n1 < S; ++n1) {
      const int idx = (k * n1) & 63;
      const float wr = twc[idx], wi = tws[idx];
      const float tr = sTr[n1 * S + m], ti = sTi[n1 * S + m];
      ar = fmaf(wr, tr, fmaf(-wi, ti, ar));
      ai = fmaf(wr, ti, fmaf(wi, tr, ai));
    }
    const float scale = 1.0f / (float)SS;
    const size_t o = (size_t)b * SS + m * S + k;
    Frt[o] = (_Float16)(ar * scale);
    Fit[o] = (_Float16)(ai * scale);
    Fst[o] = (_Float16)((ar + ai) * scale);
  }
}

// ---------------------------------------------------------------------------
// Kernel 2: per wave: 16 points; out[p] = Re[ E1(p)^T F E2(p) ]
// grid = (32 chunks, 32 batches), block = 256 (8 waves), 4 tiles per wave.
// ---------------------------------------------------------------------------
__global__ __launch_bounds__(256) void FourierInterp_eval_kernel(
    const _Float16* __restrict__ Frt, const _Float16* __restrict__ Fit,
    const _Float16* __restrict__ Fst, const float* __restrict__ xnew,
    float* __restrict__ out) {
  const int b = blockIdx.y;
  const int wave = threadIdx.x >> 5;
  const int lane = threadIdx.x & 31;
  const int half = lane >> 4;   // K-half / row-half selector
  const int col  = lane & 15;   // N column / point id within tile

  __shared__ _Float16 sFr[SS];  // F^T real,      [m][k], 8 KB
  __shared__ _Float16 sFi[SS];  // F^T imag,      8 KB
  __shared__ _Float16 sFs[SS];  // F^T real+imag, 8 KB

  // Stage the three F^T planes into LDS with CDNA5 async copies (ASYNCcnt).
  {
    const unsigned lr = (unsigned)(uintptr_t)(&sFr[0]);
    const unsigned li = (unsigned)(uintptr_t)(&sFi[0]);
    const unsigned ls = (unsigned)(uintptr_t)(&sFs[0]);
    const unsigned long long gr =
        (unsigned long long)(uintptr_t)(Frt + (size_t)b * SS);
    const unsigned long long gi =
        (unsigned long long)(uintptr_t)(Fit + (size_t)b * SS);
    const unsigned long long gs =
        (unsigned long long)(uintptr_t)(Fst + (size_t)b * SS);
#pragma unroll
    for (int j = 0; j < 2; ++j) {
      const unsigned off = (threadIdx.x + j * 256) * 16;
      asm volatile("global_load_async_to_lds_b128 %0, %1, %2"
                   :: "v"(lr + off), "v"(off), "s"(gr) : "memory");
      asm volatile("global_load_async_to_lds_b128 %0, %1, %2"
                   :: "v"(li + off), "v"(off), "s"(gi) : "memory");
      asm volatile("global_load_async_to_lds_b128 %0, %1, %2"
                   :: "v"(ls + off), "v"(off), "s"(gs) : "memory");
    }
    asm volatile("s_wait_asynccnt 0x0" ::: "memory");
  }
  __syncthreads();

#pragma unroll 1
  for (int tloop = 0; tloop < 4; ++tloop) {
    const int tile  = (blockIdx.x * 8 + wave) * 4 + tloop; // 0..1023
    const int pbase = tile * 16;

    // Coordinates of this lane's point (lanes 16-31 duplicate 0-15).
    const size_t cbase = ((size_t)b * P + pbase + col) * 2;
    const float x1 = xnew[cbase + 0];
    const float x2 = xnew[cbase + 1];

    // ---- E1 A-fragments via phase algebra (10 sincos instead of 32) ----
    // 16-bit A 16x32 layout: lane row M = col; element e of k-step ks holds
    // signed frequency fk = 16*half + e - 32*ks.
    float ec[16], es[16];
#pragma unroll
    for (int e = 0; e < 8; ++e) {
      const float fk = (float)(16 * half + e);
      __sincosf(TWO_PI * x1 * fk, &es[e], &ec[e]);
    }
    float c16, s16;
    __sincosf(TWO_PI * x1 * 8.0f, &s16, &c16);
#pragma unroll
    for (int e = 0; e < 8; ++e) {           // fk(e+8) = fk(e) + 8
      ec[8 + e] = ec[e] * c16 - es[e] * s16;
      es[8 + e] = ec[e] * s16 + es[e] * c16;
    }
    float c32, s32;                          // e^{-2*pi*i*32*x1}
    __sincosf(TWO_PI * x1 * 32.0f, &s32, &c32);
    s32 = -s32;

    v16h c1[2], s1[2], cs1[2];
#pragma unroll
    for (int e = 0; e < 16; ++e) {
      c1[0][e] = (_Float16)ec[e];
      s1[0][e] = (_Float16)es[e];
      const float cr = ec[e] * c32 - es[e] * s32;   // fk(ks=1) = fk(ks=0)-32
      const float sr = ec[e] * s32 + es[e] * c32;
      c1[1][e] = (_Float16)cr;
      s1[1][e] = (_Float16)sr;
    }
    cs1[0] = c1[0] + s1[0];                  // packed f16 adds
    cs1[1] = c1[1] + s1[1];

    // ---- E2 phases: start at fm = col-32, step +16 per N-tile ----
    float pc[8], ps[8], stc[8], sts[8], red[8];
#pragma unroll
    for (int r = 0; r < 8; ++r) {
      const float x2m = __shfl(x2, 8 * half + r, 32); // x2 of row M=8*half+r
      __sincosf(TWO_PI * x2m * (float)(col - 32), &ps[r], &pc[r]);
      __sincosf(TWO_PI * x2m * 16.0f, &sts[r], &stc[r]);
      red[r] = 0.f;
    }

#pragma unroll
    for (int idx = 0; idx < 4; ++idx) {
      const int nt = (idx + 2) & 3;  // fm sequence: col-32, col-16, col, col+16
      v8f t1 = {};
      v8f t2 = {};
      v8f t3 = {};
#pragma unroll
      for (int ks = 0; ks < 2; ++ks) {
        // B 32x16 f16 layout: lane (col + 16*half) holds column `col`,
        // K = ks*32 + 16*half + [0..15], contiguous in k within F^T.
        const int boff = (nt * 16 + col) * S + ks * 32 + 16 * half;
        const v16h br = *(const v16h*)&sFr[boff];
        const v16h bi = *(const v16h*)&sFi[boff];
        const v16h bs = *(const v16h*)&sFs[boff];
        t1 = __builtin_amdgcn_wmma_f32_16x16x32_f16(
            false, c1[ks], false, br, (short)0, t1, false, false);
        t2 = __builtin_amdgcn_wmma_f32_16x16x32_f16(
            false, s1[ks], false, bi, (short)0, t2, false, false);
        t3 = __builtin_amdgcn_wmma_f32_16x16x32_f16(
            false, cs1[ks], false, bs, (short)0, t3, false, false);
      }

      // Karatsuba recombine + E2 phase dot.
      // C/D layout: VGPR r, this lane: M = r + 8*half, N = nt*16 + col.
#pragma unroll
      for (int r = 0; r < 8; ++r) {
        const float gr = t1[r] - t2[r];            // Re(G)
        const float gi = t3[r] - t1[r] - t2[r];    // Im(G)
        red[r] += gr * pc[r] - gi * ps[r];
      }
      if (idx < 3) {
#pragma unroll
        for (int r = 0; r < 8; ++r) {       // advance phase by +16 frequency
          const float c = pc[r], s = ps[r];
          pc[r] = c * stc[r] - s * sts[r];
          ps[r] = c * sts[r] + s * stc[r];
        }
      }
    }

    // Butterfly-sum over the 16 lanes of each half (sums all 64 m values).
#pragma unroll
    for (int mask = 1; mask < 16; mask <<= 1) {
#pragma unroll
      for (int r = 0; r < 8; ++r) red[r] += __shfl_xor(red[r], mask, 32);
    }

    if (col == 0) {
      float* o = out + (size_t)b * P + pbase + 8 * half;
#pragma unroll
      for (int r = 0; r < 8; ++r) o[r] = red[r];
    }
  }
}

// ---------------------------------------------------------------------------
extern "C" void kernel_launch(void* const* d_in, const int* in_sizes, int n_in,
                              void* d_out, int out_size, void* d_ws,
                              size_t ws_size, hipStream_t stream) {
  const float* y    = (const float*)d_in[0];   // [32,64,64]
  const float* xnew = (const float*)d_in[1];   // [32,128,128,2]
  float* out        = (float*)d_out;           // [32,128,128]

  _Float16* Frt = (_Float16*)d_ws;             // [32][64][64] f16 (transposed)
  _Float16* Fit = Frt + (size_t)NB * SS;
  _Float16* Fst = Fit + (size_t)NB * SS;       // needs 768 KB total workspace

  FourierInterp_dft_kernel<<<NB, 256, 0, stream>>>(y, Frt, Fit, Fst);
  FourierInterp_eval_kernel<<<dim3(32, NB), 256, 0, stream>>>(Frt, Fit, Fst,
                                                              xnew, out);
}
